// SelfAttentionBlock_76665166233691
// MI455X (gfx1250) — matrix operand order repaired and checked by reading, not compiled, not run on hardware
//
#include <hip/hip_runtime.h>
#include <cstdint>

// ---------------------------------------------------------------------------
// SelfAttentionBlock for MI455X (gfx1250): GroupNorm -> QKV -> flash-attn -> proj
// All matmuls on v_wmma_f32_16x16x32_f16 (wave32); K/V tiles streamed to LDS
// with gfx1250 async global->LDS loads (ASYNCcnt), double-buffered.
// ---------------------------------------------------------------------------

typedef __attribute__((ext_vector_type(16))) _Float16 v16h;
typedef __attribute__((ext_vector_type(8)))  _Float16 h8;
typedef __attribute__((ext_vector_type(8)))  float    v8f;
typedef __attribute__((ext_vector_type(4)))  float    f4;
typedef __attribute__((ext_vector_type(4)))  int      v4i;

#define WMMA_F16(a, b, c) \
  __builtin_amdgcn_wmma_f32_16x16x32_f16(false, (a), false, (b), (short)0, (c), false, false)

#if defined(__has_builtin)
#if __has_builtin(__builtin_amdgcn_global_load_async_to_lds_b128) && \
    __has_builtin(__builtin_amdgcn_s_wait_asynccnt)
#define HAVE_ASYNC_COPY 1
#endif
#endif
#ifndef HAVE_ASYNC_COPY
#define HAVE_ASYNC_COPY 0
#endif

#define GLOBAL_AS __attribute__((address_space(1)))
#define LDS_AS    __attribute__((address_space(3)))

// 16B global -> LDS copy; async (ASYNCcnt-tracked) when the builtin exists.
// Builtin signature (from clang diag): (v4i AS1*, v4i AS3*, imm offset, imm cpol).
__device__ __forceinline__ void async_ld_b128(_Float16* dst_lds,
                                              const _Float16* src) {
#if HAVE_ASYNC_COPY
  GLOBAL_AS v4i* g =
      reinterpret_cast<GLOBAL_AS v4i*>(reinterpret_cast<uintptr_t>(src));
  LDS_AS v4i* l =
      reinterpret_cast<LDS_AS v4i*>(reinterpret_cast<uintptr_t>(dst_lds));
  __builtin_amdgcn_global_load_async_to_lds_b128(g, l, 0, 0);
#else
  *(h8*)dst_lds = *(const h8*)src;
#endif
}

__device__ __forceinline__ void wait_async_done() {
#if HAVE_ASYNC_COPY
  __builtin_amdgcn_s_wait_asynccnt(0);
#endif
}

// A fragment (16x32, MxK), 16-bit: lane m = lane&15, element e<8 -> K = kb+8h+e,
// e>=8 -> K = kb+16+8h+(e-8).  Two contiguous 16B chunks from row-major [M][K].
__device__ __forceinline__ v16h frag_a(const _Float16* buf, int ld, int row_base,
                                       int k_base, int lane) {
  const int h = lane >> 4, m = lane & 15;
  const _Float16* p = buf + (size_t)(row_base + m) * ld + k_base + h * 8;
  union { struct { h8 lo, hi; } s; v16h v; } u;
  u.s.lo = *(const h8*)(p);
  u.s.hi = *(const h8*)(p + 16);
  return u.v;
}

// B fragment (32x16, KxN), 16-bit: lane n = lane&15, element e<8 -> K = kb+8h+e,
// e>=8 -> K = kb+16+8h+(e-8).  Buffer layout buf[n][K] (B^T row-major): two 16B chunks per lane.
__device__ __forceinline__ v16h frag_b(const _Float16* buf, int ld, int k_base,
                                       int lane) {
  const int h = lane >> 4, n = lane & 15;
  const _Float16* p = buf + (size_t)n * ld + k_base + h * 8;
  union { struct { h8 lo, hi; } s; v16h v; } u;
  u.s.lo = *(const h8*)(p);
  u.s.hi = *(const h8*)(p + 16);
  return u.v;
}

// ---------------------------------------------------------------------------
// Kernel 0: convert weights to f16
// ---------------------------------------------------------------------------
__global__ void k_cvt_w(const float* __restrict__ qw, const float* __restrict__ pw,
                        _Float16* __restrict__ qwh, _Float16* __restrict__ pwh) {
  int i = blockIdx.x * 256 + threadIdx.x;
  if (i < 384 * 128) qwh[i] = (_Float16)qw[i];
  if (i < 128 * 128) pwh[i] = (_Float16)pw[i];
}

// ---------------------------------------------------------------------------
// Kernel 1: GroupNorm.  One block per (b, group).  Writes:
//   xn   f32 [b][c][n]  (feature-major, residual)
//   xn_t f16 [b][n][c]  (token-major: B^T operand for QKV)
// ---------------------------------------------------------------------------
__global__ __launch_bounds__(256) void k_groupnorm(
    const float* __restrict__ x, const float* __restrict__ w,
    const float* __restrict__ bia, float* __restrict__ xn,
    _Float16* __restrict__ xn_t) {
  const int N = 4096, CG = 16;
  const int b = blockIdx.x >> 3, g = blockIdx.x & 7;
  const float* xp = x + ((size_t)(b * 128 + g * CG)) * N;

  float s = 0.f, s2 = 0.f;
  for (int i = threadIdx.x; i < CG * N; i += 256) {
    float v = xp[i];
    s += v; s2 += v * v;
  }
  #pragma unroll
  for (int m = 1; m < 32; m <<= 1) {
    s  += __shfl_xor(s,  m, 32);
    s2 += __shfl_xor(s2, m, 32);
  }
  __shared__ float red[16];
  __shared__ float stat[2];
  int wid = threadIdx.x >> 5;
  if ((threadIdx.x & 31) == 0) { red[wid] = s; red[8 + wid] = s2; }
  __syncthreads();
  if (threadIdx.x == 0) {
    float ts = 0.f, ts2 = 0.f;
    for (int i = 0; i < 8; i++) { ts += red[i]; ts2 += red[8 + i]; }
    float mean = ts * (1.f / (CG * N));
    float var  = ts2 * (1.f / (CG * N)) - mean * mean;
    stat[0] = mean;
    stat[1] = rsqrtf(var + 1e-5f);
  }
  __syncthreads();
  const float mean = stat[0], inv = stat[1];
  for (int i = threadIdx.x; i < CG * N; i += 256) {
    int cc = i >> 12, n = i & 4095;
    int c = g * CG + cc;
    float v = (xp[i] - mean) * inv * w[c] + bia[c];
    xn[((size_t)(b * 128 + c)) * N + n] = v;
    xn_t[((size_t)b * N + n) * 128 + c] = (_Float16)v;
  }
}

// ---------------------------------------------------------------------------
// Kernel 2: QKV GEMM.  A = weights (M = out-chan), B^T = xn_t (N = token).
// C-tile: lane = token, vgpr row = out-chan -> q/k/v split is uniform per tile
// and q/k epilogues are a single packed 16B store per lane.
// ---------------------------------------------------------------------------
__global__ __launch_bounds__(128) void k_qkv(
    const _Float16* __restrict__ xn_t, const _Float16* __restrict__ qwh,
    const float* __restrict__ qkv_b, _Float16* __restrict__ q_t,
    _Float16* __restrict__ k_t, _Float16* __restrict__ v_f) {
  const int C = 128, N = 4096;
  const int wave = threadIdx.x >> 5, lane = threadIdx.x & 31;
  const int tile = blockIdx.x * 4 + wave;
  const int b = tile / (24 * 256);
  const int r = tile % (24 * 256);
  const int ot = r >> 8, nt = r & 255;

  const _Float16* Aw = qwh + (size_t)ot * 16 * C;              // [o][c]
  const _Float16* Bx = xn_t + ((size_t)b * N + nt * 16) * C;   // [tok][c]

  v8f acc = {};
  #pragma unroll
  for (int kc = 0; kc < 128; kc += 32) {
    v16h a  = frag_a(Aw, C, 0, kc, lane);
    v16h bb = frag_b(Bx, C, kc, lane);
    acc = WMMA_F16(a, bb, acc);
  }
  const int h = lane >> 4;
  const int tok = nt * 16 + (lane & 15);
  const int obase = ot * 16 + 8 * h;        // this lane's 8 consecutive outputs
  f4 b0 = *(const f4*)(qkv_b + obase);
  f4 b1 = *(const f4*)(qkv_b + obase + 4);
  float bv[8] = {b0.x, b0.y, b0.z, b0.w, b1.x, b1.y, b1.z, b1.w};
  const float scale = 0.08838834764831845f;  // 1/sqrt(128)

  if (ot < 8) {                 // Q: packed, scaled
    h8 st;
    #pragma unroll
    for (int v = 0; v < 8; v++) st[v] = (_Float16)((acc[v] + bv[v]) * scale);
    *(h8*)(q_t + ((size_t)b * N + tok) * C + obase) = st;
  } else if (ot < 16) {         // K: packed
    h8 st;
    #pragma unroll
    for (int v = 0; v < 8; v++) st[v] = (_Float16)(acc[v] + bv[v]);
    *(h8*)(k_t + ((size_t)b * N + tok) * C + (obase - 128)) = st;
  } else {                      // V: feature-major (strided b16)
    #pragma unroll
    for (int v = 0; v < 8; v++)
      v_f[((size_t)b * C + (obase - 256 + v)) * N + tok] =
          (_Float16)(acc[v] + bv[v]);
  }
}

// ---------------------------------------------------------------------------
// Kernel 3: flash attention.  Block = 4 waves, 64-query tile (16 rows/wave).
// Double-buffered async global->LDS staging of 64-token K/V tiles:
//   wait_asynccnt(0) -> barrier -> issue loads(j+1) -> compute(j)
// LDS: 2*16KB (K) + 2*16KB (V) + 8KB (P) = 72KB.
// ---------------------------------------------------------------------------
__global__ __launch_bounds__(128) void k_attn(
    const _Float16* __restrict__ q_t, const _Float16* __restrict__ k_t,
    const _Float16* __restrict__ v_f, _Float16* __restrict__ out_t) {
  const int C = 128, N = 4096;
  const int b = blockIdx.x >> 6, qt = blockIdx.x & 63;
  const int qbase = qt * 64;
  const int wave = threadIdx.x >> 5, lane = threadIdx.x & 31;
  const int h = lane >> 4, nloc = lane & 15;

  __shared__ __align__(16) _Float16 k_lds[2][64 * 128];  // token-major [tok][c]
  __shared__ __align__(16) _Float16 v_lds[2][128 * 64];  // feature-major [c][tok]
  __shared__ __align__(16) _Float16 p_lds[4 * 16 * 64];

  // Q fragments stay in registers (token-major, pre-scaled).
  const _Float16* qg = q_t + ((size_t)b * N + qbase + wave * 16) * C;
  v16h qa[4];
  #pragma unroll
  for (int kq = 0; kq < 4; kq++) qa[kq] = frag_a(qg, C, 0, kq * 32, lane);

  auto load_tiles = [&](int j, int buf) {
    const int jbase = j * 64;
    const _Float16* kg = k_t + ((size_t)b * N + jbase) * C;
    _Float16* kd = k_lds[buf];
    for (int i = threadIdx.x; i < 1024; i += 128)
      async_ld_b128(kd + i * 8, kg + i * 8);
    _Float16* vd = v_lds[buf];
    for (int i = threadIdx.x; i < 1024; i += 128) {
      int c = i >> 3, t8 = i & 7;
      async_ld_b128(vd + i * 8,
                    v_f + ((size_t)b * C + c) * N + jbase + t8 * 8);
    }
  };

  v8f oacc[8] = {};
  float row_m[8], row_l[8];
  #pragma unroll
  for (int v = 0; v < 8; v++) { row_m[v] = -3.0e38f; row_l[v] = 0.f; }

  _Float16* pw = p_lds + wave * 16 * 64;

  load_tiles(0, 0);  // prologue

  for (int j = 0; j < 64; j++) {
    const int cur = j & 1;
    wait_async_done();   // own async loads (tile j) landed in LDS
    __syncthreads();     // all waves' loads landed; prev compute finished
    if (j + 1 < 64) load_tiles(j + 1, cur ^ 1);  // overlap with compute below

    // S = Q K^T  (16 x 64, f32) : 16 WMMA
    v8f s[4] = {};
    #pragma unroll
    for (int kc = 0; kc < 128; kc += 32) {
      v16h a = qa[kc >> 5];
      #pragma unroll
      for (int t = 0; t < 4; t++) {
        v16h bb = frag_b(k_lds[cur] + t * 16 * 128, 128, kc, lane);
        s[t] = WMMA_F16(a, bb, s[t]);
      }
    }

    // Online softmax over 64 columns; rows are (vgpr + 8*half).
    #pragma unroll
    for (int v = 0; v < 8; v++) {
      float rv = fmaxf(fmaxf(s[0][v], s[1][v]), fmaxf(s[2][v], s[3][v]));
      rv = fmaxf(rv, __shfl_xor(rv, 1, 32));
      rv = fmaxf(rv, __shfl_xor(rv, 2, 32));
      rv = fmaxf(rv, __shfl_xor(rv, 4, 32));
      rv = fmaxf(rv, __shfl_xor(rv, 8, 32));
      float mnew  = fmaxf(row_m[v], rv);
      float alpha = __expf(row_m[v] - mnew);
      row_m[v] = mnew;
      float psum = 0.f;
      #pragma unroll
      for (int t = 0; t < 4; t++) {
        float p = __expf(s[t][v] - mnew);
        psum += p;
        pw[(v + 8 * h) * 64 + t * 16 + nloc] = (_Float16)p;
      }
      psum += __shfl_xor(psum, 1, 32);
      psum += __shfl_xor(psum, 2, 32);
      psum += __shfl_xor(psum, 4, 32);
      psum += __shfl_xor(psum, 8, 32);
      row_l[v] = row_l[v] * alpha + psum;
      #pragma unroll
      for (int t2 = 0; t2 < 8; t2++) oacc[t2][v] *= alpha;
    }

    // O += P V  (16 x 128, K=64) : 16 WMMA.  P region is private per wave.
    #pragma unroll
    for (int kc = 0; kc < 64; kc += 32) {
      v16h a = frag_a(pw, 64, 0, kc, lane);
      #pragma unroll
      for (int t2 = 0; t2 < 8; t2++) {
        v16h bb = frag_b(v_lds[cur] + t2 * 16 * 64, 64, kc, lane);
        oacc[t2] = WMMA_F16(a, bb, oacc[t2]);
      }
    }
  }

  // Normalize and write O token-major f16 [b][tok][c].
  #pragma unroll
  for (int v = 0; v < 8; v++) row_l[v] = 1.0f / row_l[v];
  const int tokbase = qbase + wave * 16;
  #pragma unroll
  for (int t2 = 0; t2 < 8; t2++) {
    #pragma unroll
    for (int v = 0; v < 8; v++) {
      int tok = tokbase + v + 8 * h;
      out_t[((size_t)b * N + tok) * C + t2 * 16 + nloc] =
          (_Float16)(oacc[t2][v] * row_l[v]);
    }
  }
}

// ---------------------------------------------------------------------------
// Kernel 4: proj GEMM + bias + residual (xn), f32 output [b][c][n].
// M = token (A = attn token-major), N = out-chan: lane = o, vgprs = 8
// consecutive tokens -> vectorized f32 residual loads and stores.
// ---------------------------------------------------------------------------
__global__ __launch_bounds__(128) void k_proj(
    const _Float16* __restrict__ out_t, const _Float16* __restrict__ pwh,
    const float* __restrict__ proj_b, const float* __restrict__ xn,
    float* __restrict__ y) {
  const int C = 128, N = 4096;
  const int wave = threadIdx.x >> 5, lane = threadIdx.x & 31;
  const int tile = blockIdx.x * 4 + wave;
  const int b = tile / (8 * 256);
  const int r = tile % (8 * 256);
  const int ot = r >> 8, nt = r & 255;

  const _Float16* A  = out_t + (size_t)b * N * C;   // [tok][c]
  const _Float16* Bw = pwh + (size_t)ot * 16 * C;   // [o][c] == B^T rows

  v8f acc = {};
  #pragma unroll
  for (int kc = 0; kc < 128; kc += 32) {
    v16h a  = frag_a(A,  C, nt * 16, kc, lane);
    v16h bb = frag_b(Bw, C, kc, lane);
    acc = WMMA_F16(a, bb, acc);
  }
  const int h = lane >> 4;
  const int o = ot * 16 + (lane & 15);
  const float bval = proj_b[o];
  const size_t base = ((size_t)b * C + o) * N + nt * 16 + 8 * h;  // 8 tokens
  f4 r0 = *(const f4*)(xn + base);
  f4 r1 = *(const f4*)(xn + base + 4);
  f4 w0, w1;
  #pragma unroll
  for (int v = 0; v < 4; v++) {
    w0[v] = acc[v]     + bval + r0[v];
    w1[v] = acc[v + 4] + bval + r1[v];
  }
  *(f4*)(y + base)     = w0;
  *(f4*)(y + base + 4) = w1;
}

// ---------------------------------------------------------------------------
extern "C" void kernel_launch(void* const* d_in, const int* in_sizes, int n_in,
                              void* d_out, int out_size, void* d_ws, size_t ws_size,
                              hipStream_t stream) {
  (void)in_sizes; (void)n_in; (void)out_size; (void)ws_size;
  const float* x      = (const float*)d_in[0];
  const float* gn_w   = (const float*)d_in[1];
  const float* gn_b   = (const float*)d_in[2];
  const float* qkv_w  = (const float*)d_in[3];
  const float* qkv_b  = (const float*)d_in[4];
  const float* proj_w = (const float*)d_in[5];
  const float* proj_b = (const float*)d_in[6];
  float* out = (float*)d_out;

  const size_t B = 8, C = 128, N = 4096;
  char* ws = (char*)d_ws;
  size_t off = 0;
  float*    xn    = (float*)(ws + off);    off += B * C * N * 4;   // 16 MB
  _Float16* xn_t  = (_Float16*)(ws + off); off += B * N * C * 2;   // 8 MB
  _Float16* q_t   = (_Float16*)(ws + off); off += B * N * C * 2;
  _Float16* k_t   = (_Float16*)(ws + off); off += B * N * C * 2;
  _Float16* v_f   = (_Float16*)(ws + off); off += B * C * N * 2;
  _Float16* attn  = (_Float16*)(ws + off); off += B * N * C * 2;
  _Float16* qwh   = (_Float16*)(ws + off); off += 384 * 128 * 2;
  _Float16* pwh   = (_Float16*)(ws + off); off += 128 * 128 * 2;

  k_cvt_w<<<192, 256, 0, stream>>>(qkv_w, proj_w, qwh, pwh);
  k_groupnorm<<<64, 256, 0, stream>>>(x, gn_w, gn_b, xn, xn_t);
  k_qkv<<<12288, 128, 0, stream>>>(xn_t, qwh, qkv_b, q_t, k_t, v_f);
  k_attn<<<512, 128, 0, stream>>>(q_t, k_t, v_f, attn);
  k_proj<<<4096, 128, 0, stream>>>(attn, pwh, proj_b, xn, out);
}
